// ModulatedConv2d_18502719111885
// MI455X (gfx1250) — matrix operand order repaired and checked by reading, not compiled, not run on hardware
//
#include <hip/hip_runtime.h>

// ---------------------------------------------------------------------------
// ModulatedConv2d (StyleGAN2) for gfx1250 / MI455X
//   x[8,128,128,128] f32, latent[8,512], weight[128,128,3,3], bias[128],
//   mlp_w[128,512], mlp_b[128]  ->  out[8,128,128,128] f32
// Conv = per-sample implicit GEMM, M=128 (O), N=16384 (pixels), K=1152,
// executed with V_WMMA_F32_16X16X32_F16, B tiles staged to LDS with
// global_load_async_to_lds_b128 (TENSOR/ASYNC data path, wave32).
// ---------------------------------------------------------------------------

typedef _Float16 v16h __attribute__((ext_vector_type(16)));
typedef _Float16 v8h  __attribute__((ext_vector_type(8)));
typedef float    v8f  __attribute__((ext_vector_type(8)));

#define BATCH   8
#define FIN     128
#define FOUT    128
#define KSZ     3
#define LATENT  512
#define HH      128
#define WW      128
#define KTOT    (FIN * KSZ * KSZ)   // 1152, GEMM K, order k = (ky*3+kx)*128 + i
#define EPSV    1e-5f

// Toggle for the CDNA5 async-to-LDS staging path (inline asm, VFLAT encoding).
#define USE_ASYNC 1

// ---------------------------------------------------------------------------
// Kernel 1: s[b,i] = sum_l latent[b,l] * mlp_w[i,l] * sqrt(2/512) + mlp_b[i]
// sqrt(2/512) == 1/16 exactly.
// ---------------------------------------------------------------------------
__global__ __launch_bounds__(256) void mc_compute_s(
    const float* __restrict__ latent, const float* __restrict__ mlp_w,
    const float* __restrict__ mlp_b, float* __restrict__ s) {
  int idx = blockIdx.x * 256 + threadIdx.x;      // 0..1023
  int b = idx >> 7;
  int i = idx & 127;
  const float* lb = latent + (size_t)b * LATENT;
  const float* wr = mlp_w + (size_t)i * LATENT;
  float acc = 0.f;
#pragma unroll 8
  for (int l = 0; l < LATENT; ++l) acc += lb[l] * wr[l];
  s[idx] = acc * 0.0625f + mlp_b[i];
}

// ---------------------------------------------------------------------------
// Kernel 2: per (b,o): v[e] = s[b,i] * weight[o,i,ky,kx] * (1/24)
//           d = rsqrt(sum v^2 + eps);  wmod[b][o][e] = f16(v[e] * d)
// wmod stored in GEMM k-order: e = (ky*3+kx)*128 + i.  sqrt(2/1152)==1/24.
// ---------------------------------------------------------------------------
__global__ __launch_bounds__(256) void mc_modulate(
    const float* __restrict__ s, const float* __restrict__ weight,
    _Float16* __restrict__ wmod) {
  const int o = blockIdx.x;
  const int b = blockIdx.y;
  const int tid = threadIdx.x;
  __shared__ float red[256];

  const float wscale = 1.0f / 24.0f;
  float v[5];
  float acc = 0.f;
#pragma unroll
  for (int it = 0; it < 5; ++it) {
    int e = tid + it * 256;
    float val = 0.f;
    if (e < KTOT) {
      int i = e & 127;          // input channel
      int kyx = e >> 7;         // 0..8 = ky*3+kx
      val = s[b * FIN + i] * weight[((size_t)o * FIN + i) * 9 + kyx] * wscale;
    }
    v[it] = val;
    acc += val * val;
  }
  red[tid] = acc;
  __syncthreads();
#pragma unroll
  for (int off = 128; off > 0; off >>= 1) {
    if (tid < off) red[tid] += red[tid + off];
    __syncthreads();
  }
  float d = rsqrtf(red[0] + EPSV);
  _Float16* dst = wmod + ((size_t)b * FOUT + o) * KTOT;
#pragma unroll
  for (int it = 0; it < 5; ++it) {
    int e = tid + it * 256;
    if (e < KTOT) dst[e] = (_Float16)(v[it] * d);
  }
}

// ---------------------------------------------------------------------------
// Kernel 2.5: layout transform x NCHW f32 -> xh NHWC f16 (xh[b][y][x][i]).
// LDS-tiled so both global read and global write are coalesced.
// Grid: (W/32, H, B); block 256.  Tile = 32 x-positions * 128 channels.
// ---------------------------------------------------------------------------
__global__ __launch_bounds__(256) void mc_nchw2nhwc(
    const float* __restrict__ x, _Float16* __restrict__ xh) {
  const int x0 = blockIdx.x * 32;
  const int y  = blockIdx.y;
  const int b  = blockIdx.z;
  const int tid = threadIdx.x;
  __shared__ _Float16 T[32][FIN + 4];

#pragma unroll
  for (int r = 0; r < 16; ++r) {               // 4096 elems, i-major read
    int idx = tid + r * 256;
    int i  = idx >> 5;                          // 0..127
    int xx = idx & 31;                          // coalesced in x
    T[xx][i] = (_Float16)x[(((size_t)b * FIN + i) * HH + y) * WW + x0 + xx];
  }
  __syncthreads();
#pragma unroll
  for (int r = 0; r < 16; ++r) {               // x-major write, i coalesced
    int idx = tid + r * 256;
    int xx = idx >> 7;                          // 0..31
    int i  = idx & 127;
    xh[(((size_t)b * HH + y) * WW + x0 + xx) * FIN + i] = T[xx][i];
  }
}

// ---------------------------------------------------------------------------
// Kernel 3 (main): implicit-GEMM conv with WMMA, NHWC f16 input.
// Grid: x = h (128 rows), y = batch. Block 256 = 8 waves.
// Wave w owns M-tile w (16 out channels) and all 8 N-tiles (full 128-px row).
// Per ky tap: stage one input row (128 px * 128 ch f16 = 32KB) into a haloed
// LDS buffer via async-to-LDS, then 3(kx) * 4(chunks) * 8(N-tiles) = 96 WMMAs
// between one barrier pair. kx handled as a +/-1-row LDS offset.
// ---------------------------------------------------------------------------
__global__ __launch_bounds__(256) void mc_conv_wmma_nhwc(
    const _Float16* __restrict__ xh, const _Float16* __restrict__ wmod,
    const float* __restrict__ bias, float* __restrict__ out) {
  const int b   = blockIdx.y;
  const int h   = blockIdx.x;
  const int tid = threadIdx.x;
  const int wave = tid >> 5;      // M-tile id, 0..7
  const int lane = tid & 31;
  const int lm   = lane & 15;
  const int hv   = lane >> 4;

  constexpr int RSTR = 136;       // row stride in halves: 272B, 16B multiple
  __shared__ __align__(16) _Float16 Xs[130 * RSTR];   // rows: x = -1..128

  // Zero once: halo rows 0 and 129 plus pad columns stay zero forever.
  for (int idx = tid; idx < 130 * RSTR; idx += 256) Xs[idx] = (_Float16)0.f;

  v8f acc[8] = {};

  const _Float16* wrow = wmod + ((size_t)b * FOUT + wave * 16 + lm) * KTOT;
  const _Float16* xrow = xh + (size_t)b * HH * WW * FIN;   // [y][x][i]

  for (int ky = 0; ky < KSZ; ++ky) {
    const int y = h + ky - 1;
    __syncthreads();              // previous stage fully consumed
    if (y >= 0 && y < HH) {
      // Thread t copies 64 halves (128B): x-position t>>1, channel half t&1.
      const int xx = tid >> 1;
      const int cp = tid & 1;
      const _Float16* src = xrow + ((size_t)y * WW + xx) * FIN + cp * 64;
      _Float16* dst = &Xs[(xx + 1) * RSTR + cp * 64];
#pragma unroll
      for (int j = 0; j < 8; ++j) {
#if USE_ASYNC
        // VFLAT async load: vDst = LDS byte offset, vAddr64 = global address.
        unsigned int loff = (unsigned int)(size_t)(dst + j * 8);
        unsigned long long ga = (unsigned long long)(size_t)(src + j * 8);
        asm volatile("global_load_async_to_lds_b128 %0, %1, off"
                     :: "v"(loff), "v"(ga) : "memory");
#else
        *(v8h*)(dst + j * 8) = *(const v8h*)(src + j * 8);
#endif
      }
    } else {
      // out-of-image tap row: zero the 128 interior rows
      for (int idx = tid; idx < 128 * RSTR; idx += 256)
        Xs[RSTR + idx] = (_Float16)0.f;
    }
#if USE_ASYNC
    asm volatile("s_wait_asynccnt 0" ::: "memory");
#endif
    __syncthreads();

#pragma unroll
    for (int kx = 0; kx < KSZ; ++kx) {
#pragma unroll
      for (int c = 0; c < 4; ++c) {
        const int i0 = c * 32;
        const int k0 = (ky * KSZ + kx) * FIN + i0;
        // A fragment (16x32 f16): lane half hv holds K = hv*8 + {0..7,16..23}
        v8h alo = *(const v8h*)(wrow + k0 + hv * 8);
        v8h ahi = *(const v8h*)(wrow + k0 + hv * 8 + 16);
        v16h a;
#pragma unroll
        for (int j = 0; j < 8; ++j) { a[j] = alo[j]; a[j + 8] = ahi[j]; }
#pragma unroll
        for (int nt = 0; nt < 8; ++nt) {
          // B element (kk,n) = x[b][i0+kk][y][n+kx-1] = Xs[n+kx][i0+kk]
          const _Float16* bp = &Xs[(nt * 16 + lm + kx) * RSTR + i0 + hv * 8];
          v8h blo = *(const v8h*)(bp);
          v8h bhi = *(const v8h*)(bp + 16);
          v16h bf;
#pragma unroll
          for (int j = 0; j < 8; ++j) { bf[j] = blo[j]; bf[j + 8] = bhi[j]; }
          acc[nt] = __builtin_amdgcn_wmma_f32_16x16x32_f16(
              false, a, false, bf, (short)0, acc[nt], false, false);
        }
      }
    }
  }

  // Epilogue: D layout lane(lm,hv), VGPR r -> M = r + hv*8.
#pragma unroll
  for (int nt = 0; nt < 8; ++nt) {
    const int w = nt * 16 + lm;
#pragma unroll
    for (int r = 0; r < 8; ++r) {
      const int o = wave * 16 + hv * 8 + r;
      out[(((size_t)b * FOUT + o) * HH + h) * WW + w] = acc[nt][r] + bias[o];
    }
  }
}

// ---------------------------------------------------------------------------
// Fallback conv (round-0 version, small workspace): f32 loads + in-loop cvt.
// ---------------------------------------------------------------------------
__global__ __launch_bounds__(256) void mc_conv_wmma_small(
    const float* __restrict__ x, const _Float16* __restrict__ wmod,
    const float* __restrict__ bias, float* __restrict__ out) {
  const int b  = blockIdx.y;
  const int h  = blockIdx.x >> 1;
  const int w0 = (blockIdx.x & 1) * 64;
  const int tid  = threadIdx.x;
  const int wave = tid >> 5;
  const int lane = tid & 31;
  const int lm   = lane & 15;
  const int hv   = lane >> 4;

  __shared__ __align__(16) _Float16 Xs[64][40];
  v8f acc[4] = {};
  const _Float16* wrow = wmod + ((size_t)b * FOUT + wave * 16 + lm) * KTOT;

  for (int kstep = 0; kstep < KTOT / 32; ++kstep) {
    const int k0  = kstep * 32;
    const int kyx = k0 >> 7;
    const int i0  = k0 & 127;
    const int ky  = kyx / 3;
    const int kx  = kyx - ky * 3;
    {
      const int n   = tid & 63;
      const int grp = tid >> 6;
      const int y   = h + ky - 1;
      const int xg  = w0 + n + kx - 1;
      const bool inb = (y >= 0) && (y < HH) && (xg >= 0) && (xg < WW);
#pragma unroll
      for (int j = 0; j < 8; ++j) {
        int kk = grp * 8 + j;
        float v = 0.f;
        if (inb) v = x[(((size_t)b * FIN + i0 + kk) * HH + y) * WW + xg];
        Xs[n][kk] = (_Float16)v;
      }
    }
    __syncthreads();
    v8h alo = *(const v8h*)(wrow + k0 + hv * 8);
    v8h ahi = *(const v8h*)(wrow + k0 + hv * 8 + 16);
    v16h a;
#pragma unroll
    for (int j = 0; j < 8; ++j) { a[j] = alo[j]; a[j + 8] = ahi[j]; }
#pragma unroll
    for (int nt = 0; nt < 4; ++nt) {
      const _Float16* bp = &Xs[nt * 16 + lm][hv * 8];
      v8h blo = *(const v8h*)(bp);
      v8h bhi = *(const v8h*)(bp + 16);
      v16h bf;
#pragma unroll
      for (int j = 0; j < 8; ++j) { bf[j] = blo[j]; bf[j + 8] = bhi[j]; }
      acc[nt] = __builtin_amdgcn_wmma_f32_16x16x32_f16(
          false, a, false, bf, (short)0, acc[nt], false, false);
    }
    __syncthreads();
  }
#pragma unroll
  for (int nt = 0; nt < 4; ++nt) {
    const int w = w0 + nt * 16 + lm;
#pragma unroll
    for (int r = 0; r < 8; ++r) {
      const int o = wave * 16 + hv * 8 + r;
      out[(((size_t)b * FOUT + o) * HH + h) * WW + w] = acc[nt][r] + bias[o];
    }
  }
}

// ---------------------------------------------------------------------------
extern "C" void kernel_launch(void* const* d_in, const int* in_sizes, int n_in,
                              void* d_out, int out_size, void* d_ws, size_t ws_size,
                              hipStream_t stream) {
  const float* x      = (const float*)d_in[0];
  const float* latent = (const float*)d_in[1];
  const float* weight = (const float*)d_in[2];
  const float* bias   = (const float*)d_in[3];
  const float* mlp_w  = (const float*)d_in[4];
  const float* mlp_b  = (const float*)d_in[5];
  float* out = (float*)d_out;

  // workspace: s[1024] f32 | wmod[8*128*1152] f16 | xh[8*128*128*128] f16
  const size_t S_OFF    = 0;
  const size_t WMOD_OFF = 4096;
  const size_t XH_OFF   = WMOD_OFF + (size_t)BATCH * FOUT * KTOT * 2;   // 0x241000
  const size_t NEED_BIG = XH_OFF + (size_t)BATCH * HH * WW * FIN * 2;   // ~35.9 MB

  float*    s    = (float*)((char*)d_ws + S_OFF);
  _Float16* wmod = (_Float16*)((char*)d_ws + WMOD_OFF);
  _Float16* xh   = (_Float16*)((char*)d_ws + XH_OFF);

  mc_compute_s<<<dim3(4), dim3(256), 0, stream>>>(latent, mlp_w, mlp_b, s);
  mc_modulate<<<dim3(FOUT, BATCH), dim3(256), 0, stream>>>(s, weight, wmod);

  if (ws_size >= NEED_BIG) {
    mc_nchw2nhwc<<<dim3(WW / 32, HH, BATCH), dim3(256), 0, stream>>>(x, xh);
    mc_conv_wmma_nhwc<<<dim3(HH, BATCH), dim3(256), 0, stream>>>(xh, wmod, bias, out);
  } else {
    mc_conv_wmma_small<<<dim3(256, BATCH), dim3(256), 0, stream>>>(x, wmod, bias, out);
  }
}
